// SelectiveMixing_85203561218267
// MI455X (gfx1250) — matrix operand order repaired and checked by reading, not compiled, not run on hardware
//
#include <hip/hip_runtime.h>

// ---------------------------------------------------------------------------
// SelectiveMixing on MI455X (gfx1250): bf16 WMMA GEMMs + fused flash attention
// Round 5: sched_group_barrier pipelining — phase-shift loads one k-step
//          ahead of the WMMAs so waits stay partial instead of drain-to-zero.
// ---------------------------------------------------------------------------

#define DIMD   1024
#define INNERD 1536
#define TDIMD  3072          // 3*DIMD
#define BATCH  4
#define SEQ    2048
#define ROWS   8192          // BATCH*SEQ

typedef __attribute__((ext_vector_type(16))) __bf16 bf16x16;
typedef __attribute__((ext_vector_type(8)))  __bf16 bf16x8;
typedef __attribute__((ext_vector_type(8)))  float  floatx8;
typedef __attribute__((ext_vector_type(4)))  unsigned int uint4v;
typedef __attribute__((ext_vector_type(8)))  int int8v;
typedef __attribute__((ext_vector_type(4)))  int int4v;

#if defined(__has_builtin)
#if __has_builtin(__builtin_amdgcn_tensor_load_to_lds) && \
    __has_builtin(__builtin_amdgcn_s_wait_tensorcnt)
#define HAVE_TDM 1
#endif
#if __has_builtin(__builtin_amdgcn_sched_group_barrier)
#define SGB(mask, n) __builtin_amdgcn_sched_group_barrier((mask), (n), 0)
#endif
#endif
#ifndef HAVE_TDM
#define HAVE_TDM 0
#endif
#ifndef SGB
#define SGB(mask, n)
#endif
// sched_group_barrier masks
#define SG_WMMA  0x0008
#define SG_VMRD  0x0020
#define SG_DSRD  0x0100

union FragU { bf16x16 v; bf16x8 h[2]; };

// A fragment: 16x32 bf16, A row-major [16][lda]; lane m = lane&15, half = lane>>4
__device__ __forceinline__ bf16x16 load_a_frag(const __bf16* A, int lda, int k0) {
    const int lane = threadIdx.x & 31;
    const int m   = lane & 15;
    const int hlf = lane >> 4;
    const __bf16* p = A + (size_t)m * lda + k0 + hlf * 8;
    FragU f;
    f.h[0] = *(const bf16x8*)(p);
    f.h[1] = *(const bf16x8*)(p + 16);
    return f.v;
}

// B fragment: 32x16 bf16 from transposed storage Bt[N][K]; lane n = lane&15
__device__ __forceinline__ bf16x16 load_b_frag(const __bf16* Bt, int ldb, int k0) {
    const int lane = threadIdx.x & 31;
    const int n   = lane & 15;
    const int hlf = lane >> 4;
    const __bf16* p = Bt + (size_t)n * ldb + k0 + hlf * 8;
    FragU f;
    f.h[0] = *(const bf16x8*)(p);
    f.h[1] = *(const bf16x8*)(p + 16);
    return f.v;
}

#define WMMA_BF16(a, b, c) \
    __builtin_amdgcn_wmma_f32_16x16x32_bf16(false, (a), false, (b), (short)0, (c), false, false)

// Software-pipelined 16x64 output tile GEMM body: A[16][KDIM] row-major,
// Bt[64 rows][KDIM] (transposed weights). Fully unrolled; sched_group_barrier
// enforces the pattern  V(10) | V(10) W(4) | V(10) W(4) | ...  so step i's
// WMMAs issue after step i+1's loads (loads stay in flight across WMMAs).
template <int KDIM>
__device__ __forceinline__ void gemm_tile_16x64(const __bf16* __restrict__ Abase,
                                                const __bf16* __restrict__ Bt0,
                                                floatx8 acc[4]) {
    constexpr int NSTEP = KDIM / 32;
    bf16x16 a_buf[2];
    bf16x16 b_buf[2][4];
    a_buf[0] = load_a_frag(Abase, KDIM, 0);
#pragma unroll
    for (int t = 0; t < 4; ++t)
        b_buf[0][t] = load_b_frag(Bt0 + (size_t)t * 16 * KDIM, KDIM, 0);
    SGB(SG_VMRD, 10);                       // phase-shift: step-0 loads first

#pragma unroll
    for (int i = 0; i < NSTEP; ++i) {
        const int cur = i & 1, nxt = cur ^ 1;
        if (i + 1 < NSTEP) {
            a_buf[nxt] = load_a_frag(Abase, KDIM, (i + 1) * 32);
#pragma unroll
            for (int t = 0; t < 4; ++t)
                b_buf[nxt][t] =
                    load_b_frag(Bt0 + (size_t)t * 16 * KDIM, KDIM, (i + 1) * 32);
        }
#pragma unroll
        for (int t = 0; t < 4; ++t)
            acc[t] = WMMA_BF16(a_buf[cur], b_buf[cur][t], acc[t]);
        SGB(SG_VMRD, 10);                   // step i+1 loads ...
        SGB(SG_WMMA, 4);                    // ... before step i WMMAs
    }
}

// ---------------------------------------------------------------------------
// Precision conversion kernels
// ---------------------------------------------------------------------------
__global__ void k_f32_to_bf16(const float* __restrict__ in, __bf16* __restrict__ out, int n) {
    int i = blockIdx.x * blockDim.x + threadIdx.x;
    if (i < n) out[i] = (__bf16)in[i];
}

// in: [K][N] fp32 row-major  ->  out: [N][K] bf16 (transposed)
__global__ void k_transpose_bf16(const float* __restrict__ in, __bf16* __restrict__ out,
                                 int K, int N) {
    int i = blockIdx.x * blockDim.x + threadIdx.x;
    if (i < K * N) {
        int k = i / N, n = i % N;
        out[(size_t)n * K + k] = (__bf16)in[i];
    }
}

// ---------------------------------------------------------------------------
// GEMM1: h = silu(x @ W1 + b1)
// ---------------------------------------------------------------------------
__global__ void __launch_bounds__(32) k_gemm1_silu(const __bf16* __restrict__ xb,
                                                   const __bf16* __restrict__ W1t,
                                                   const float*  __restrict__ b1,
                                                   __bf16* __restrict__ h) {
    const int row0 = blockIdx.x * 16;
    const int n0   = blockIdx.y * 64;
    floatx8 acc[4] = {};
    gemm_tile_16x64<DIMD>(xb + (size_t)row0 * DIMD, W1t + (size_t)n0 * DIMD, acc);

    const int lane = threadIdx.x & 31;
    const int nl = lane & 15, hlf = lane >> 4;
#pragma unroll
    for (int t = 0; t < 4; ++t) {
        const int col = n0 + t * 16 + nl;
        const float bias = b1[col];
#pragma unroll
        for (int r = 0; r < 8; ++r) {
            const int row = row0 + r + 8 * hlf;
            float v = acc[t][r] + bias;
            v = v / (1.0f + __expf(-v));          // silu
            h[(size_t)row * INNERD + col] = (__bf16)v;
        }
    }
}

// ---------------------------------------------------------------------------
// GEMM2: qkg = h @ W2 + b2 ; split into q, k, sigmoid(g)
// ---------------------------------------------------------------------------
__global__ void __launch_bounds__(32) k_gemm2_qkg(const __bf16* __restrict__ h,
                                                  const __bf16* __restrict__ W2t,
                                                  const float*  __restrict__ b2,
                                                  __bf16* __restrict__ qb,
                                                  __bf16* __restrict__ kb,
                                                  __bf16* __restrict__ sg) {
    const int row0 = blockIdx.x * 16;
    const int n0   = blockIdx.y * 64;
    floatx8 acc[4] = {};
    gemm_tile_16x64<INNERD>(h + (size_t)row0 * INNERD, W2t + (size_t)n0 * INNERD, acc);

    const int lane = threadIdx.x & 31;
    const int nl = lane & 15, hlf = lane >> 4;
#pragma unroll
    for (int t = 0; t < 4; ++t) {
        const int col = n0 + t * 16 + nl;
        const float bias = b2[col];
#pragma unroll
        for (int r = 0; r < 8; ++r) {
            const int row = row0 + r + 8 * hlf;
            float v = acc[t][r] + bias;
            if (col < DIMD) {
                qb[(size_t)row * DIMD + col] = (__bf16)v;
            } else if (col < 2 * DIMD) {
                kb[(size_t)row * DIMD + (col - DIMD)] = (__bf16)v;
            } else {
                sg[(size_t)row * DIMD + (col - 2 * DIMD)] =
                    (__bf16)(1.0f / (1.0f + __expf(-v)));   // sigmoid gate
            }
        }
    }
}

// ---------------------------------------------------------------------------
// GEMM3: v = (x @ Wv + bv) * sig(g); stored TRANSPOSED per batch: vT[b][d][l]
// ---------------------------------------------------------------------------
__global__ void __launch_bounds__(32) k_gemm3_v(const __bf16* __restrict__ xb,
                                                const __bf16* __restrict__ Wvt,
                                                const float*  __restrict__ bv,
                                                const __bf16* __restrict__ sg,
                                                __bf16* __restrict__ vT) {
    const int row0 = blockIdx.x * 16;
    const int n0   = blockIdx.y * 64;
    floatx8 acc[4] = {};
    gemm_tile_16x64<DIMD>(xb + (size_t)row0 * DIMD, Wvt + (size_t)n0 * DIMD, acc);

    const int lane = threadIdx.x & 31;
    const int nl = lane & 15, hlf = lane >> 4;
    const int bb = row0 / SEQ;
    const int l0 = (row0 % SEQ) + 8 * hlf;      // 8 consecutive sequence positions
#pragma unroll
    for (int t = 0; t < 4; ++t) {
        const int d = n0 + t * 16 + nl;
        const float bias = bv[d];
        bf16x8 pack;
#pragma unroll
        for (int r = 0; r < 8; ++r) {
            const int row = row0 + r + 8 * hlf;
            float v = (acc[t][r] + bias) * (float)sg[(size_t)row * DIMD + d];
            pack[r] = (__bf16)v;
        }
        *(bf16x8*)(vT + ((size_t)bb * DIMD + d) * SEQ + l0) = pack;   // 16B store
    }
}

// ---------------------------------------------------------------------------
// Fused flash attention + gate mixing + RMSNorm.
// Workgroup = 8 waves = 256 threads handles one 16-row query tile, full D.
// ---------------------------------------------------------------------------
__global__ void __launch_bounds__(256) k_attention(const __bf16* __restrict__ qb,
                                                   const __bf16* __restrict__ kb,
                                                   const __bf16* __restrict__ vT,
                                                   const float*  __restrict__ x,
                                                   const float*  __restrict__ mixw,
                                                   const float*  __restrict__ normw,
                                                   float* __restrict__ out) {
    __shared__ __bf16 qs[16 * DIMD];          // 32 KB query tile
    __shared__ float  Sblk[16][128];          // 8 KB raw scores
    __shared__ __bf16 Pblk[16 * 128];         // 4 KB exp() probs (bf16 for WMMA A)
    __shared__ float  mrow[16], lrow[16], arow[16], rs[16];
    __shared__ float  red[16][16];

    const int bb  = blockIdx.y;
    const int qt  = blockIdx.x;
    const int tid = threadIdx.x;
    const int wid = tid >> 5;
    const int lane = tid & 31;
    const int nl = lane & 15, hlf = lane >> 4;
    const int d0w = wid * 128;

    // ---- stage the contiguous 16xD (32 KB) query tile into LDS ----------
    const __bf16* qsrc = qb + ((size_t)bb * SEQ + qt * 16) * DIMD;
#if HAVE_TDM
    // Tensor Data Mover: 1-D descriptor, data_size=8B, 4096 units = 32 KB.
    if (wid == 0) {
        const unsigned long long ga = (unsigned long long)(uintptr_t)qsrc;
        const unsigned lds_off = (unsigned)(uintptr_t)&qs[0];   // low 32b = LDS offset
        uint4v g0;
        g0[0] = 1u;                                   // count=1, user descriptor
        g0[1] = lds_off;                              // lds_addr
        g0[2] = (unsigned)(ga & 0xffffffffu);         // global_addr[31:0]
        g0[3] = (unsigned)((ga >> 32) & 0x01ffffffu)  // global_addr[56:32]
                | (2u << 30);                         // type = 2 ("image")
        int8v g1;
        g1[0] = (3 << 16);                            // data_size = 8 bytes
        g1[1] = (int)(0x1000u << 16);                 // tensor_dim0 = 4096 (lo16)
        g1[2] = (int)(1u << 16);                      // tensor_dim0 hi=0 | tensor_dim1 = 1
        g1[3] = (int)(0x1000u << 16);                 // tile_dim0 = 4096
        g1[4] = 0;                                    // tile_dim1 = tile_dim2 = 0
        g1[5] = 0x1000;                               // tensor_dim0_stride = 4096
        g1[6] = 0;
        g1[7] = 0;
        const int4v gz4 = {0, 0, 0, 0};
        const int8v gz8 = {0, 0, 0, 0, 0, 0, 0, 0};
        // 6-arg (clang-23 / therock-10.0) form
        __builtin_amdgcn_tensor_load_to_lds(g0, g1, gz4, gz4, gz8, 0);
        __builtin_amdgcn_s_wait_tensorcnt(0);         // data in LDS before barrier
    }
#else
    for (int i = tid; i < 16 * DIMD / 8; i += 256)
        ((bf16x8*)qs)[i] = ((const bf16x8*)qsrc)[i];
#endif
    if (tid < 16) { mrow[tid] = -1e30f; lrow[tid] = 0.0f; rs[tid] = 0.0f; }
    __syncthreads();

    floatx8 o[8] = {};                        // O accumulator: 16 rows x 128 cols

    for (int kbi = 0; kbi < SEQ / 128; ++kbi) {
        // ---- S tile: this wave computes 16 keys (pipelined B fetch) -----
        const int key0 = kbi * 128 + wid * 16;
        floatx8 s = {};
        const __bf16* kbase = kb + ((size_t)bb * SEQ + key0) * DIMD;
        {
            constexpr int NS = DIMD / 32;
            bf16x16 bbuf[2];
            bbuf[0] = load_b_frag(kbase, DIMD, 0);
            SGB(SG_VMRD, 2);                  // phase-shift group
#pragma unroll
            for (int i = 0; i < NS; ++i) {
                const int cur = i & 1;
                if (i + 1 < NS)
                    bbuf[cur ^ 1] = load_b_frag(kbase, DIMD, (i + 1) * 32);
                bf16x16 a = load_a_frag(qs, DIMD, i * 32);
                s = WMMA_BF16(a, bbuf[cur], s);
                SGB(SG_VMRD, 2);              // step i+1 global loads
                SGB(SG_DSRD, 2);              // step i LDS A loads
                SGB(SG_WMMA, 1);              // step i WMMA
            }
        }
#pragma unroll
        for (int r = 0; r < 8; ++r)
            Sblk[r + 8 * hlf][wid * 16 + nl] = s[r] * 0.03125f;   // 1/sqrt(D)
        __syncthreads();

        // ---- online softmax over the 128-key block ----------------------
        const int rid = tid & 15;             // row 0..15
        const int cid = tid >> 4;             // col group 0..15 (8 cols each)
        float pm = -1e30f;
#pragma unroll
        for (int j = 0; j < 8; ++j) pm = fmaxf(pm, Sblk[rid][cid * 8 + j]);
        red[rid][cid] = pm;
        __syncthreads();
        if (tid < 16) {
            float bm = mrow[tid];
            for (int c = 0; c < 16; ++c) bm = fmaxf(bm, red[tid][c]);
            arow[tid] = __expf(mrow[tid] - bm);   // 0 on first block
            mrow[tid] = bm;
        }
        __syncthreads();
        const float bm = mrow[rid];
        float ps = 0.0f;
#pragma unroll
        for (int j = 0; j < 8; ++j) {
            float e = __expf(Sblk[rid][cid * 8 + j] - bm);
            Pblk[rid * 128 + cid * 8 + j] = (__bf16)e;
            ps += e;
        }
        red[rid][cid] = ps;
        __syncthreads();
        if (tid < 16) {
            float ssum = 0.0f;
            for (int c = 0; c < 16; ++c) ssum += red[tid][c];
            lrow[tid] = lrow[tid] * arow[tid] + ssum;
        }
        __syncthreads();

        // ---- rescale O, accumulate P @ V (pipelined vT fetch) -----------
        float al[8];
#pragma unroll
        for (int r = 0; r < 8; ++r) al[r] = arow[r + 8 * hlf];
#pragma unroll
        for (int t = 0; t < 8; ++t)
#pragma unroll
            for (int r = 0; r < 8; ++r) o[t][r] = o[t][r] * al[r];

        {
            const __bf16* vbase = vT + ((size_t)bb * DIMD + d0w) * SEQ + kbi * 128;
            bf16x16 bc = load_b_frag(vbase, SEQ, 0);      // (t=0, ks=0)
            SGB(SG_VMRD, 2);                  // phase-shift group
#pragma unroll
            for (int ks = 0; ks < 4; ++ks) {
                bf16x16 a = load_a_frag(Pblk, 128, ks * 32);
                SGB(SG_DSRD, 2);              // P-tile LDS loads for this ks
#pragma unroll
                for (int t = 0; t < 8; ++t) {
                    bf16x16 bn = bc;
                    if (!(ks == 3 && t == 7)) {           // compile-time resolved
                        const int tn  = (t + 1) & 7;
                        const int ksn = (t == 7) ? ks + 1 : ks;
                        bn = load_b_frag(vbase + (size_t)tn * 16 * SEQ, SEQ, ksn * 32);
                    }
                    o[t] = WMMA_BF16(a, bc, o[t]);
                    bc = bn;
                    SGB(SG_VMRD, 2);          // next-tile loads ...
                    SGB(SG_WMMA, 1);          // ... before current WMMA
                }
            }
        }
        __syncthreads();
    }

    // ---- finalize: 1/l, mix with x, RMSNorm ----------------------------
    float invl[8];
#pragma unroll
    for (int r = 0; r < 8; ++r) invl[r] = 1.0f / lrow[r + 8 * hlf];
    const int row0g = bb * SEQ + qt * 16;
    float ssq[8];
#pragma unroll
    for (int r = 0; r < 8; ++r) ssq[r] = 0.0f;
#pragma unroll
    for (int t = 0; t < 8; ++t) {
        const int d = d0w + t * 16 + nl;
        const float mw = mixw[d];
#pragma unroll
        for (int r = 0; r < 8; ++r) {
            const int rowg = row0g + r + 8 * hlf;
            const float g  = o[t][r] * invl[r];
            const float xv = x[(size_t)rowg * DIMD + d];
            const float y  = mw * g + (1.0f - mw) * xv;
            o[t][r] = y;
            ssq[r] += y * y;
        }
    }
    // reduce squared sums across the 16 lanes of each half
#pragma unroll
    for (int msk = 1; msk <= 8; msk <<= 1)
#pragma unroll
        for (int r = 0; r < 8; ++r) ssq[r] += __shfl_xor(ssq[r], msk, 32);
    if (nl == 0) {
#pragma unroll
        for (int r = 0; r < 8; ++r) atomicAdd(&rs[r + 8 * hlf], ssq[r]);
    }
    __syncthreads();
    float inorm[8];
#pragma unroll
    for (int r = 0; r < 8; ++r)
        inorm[r] = 1.0f / sqrtf(rs[r + 8 * hlf] * (1.0f / DIMD) + 1e-6f);
#pragma unroll
    for (int t = 0; t < 8; ++t) {
        const int d = d0w + t * 16 + nl;
        const float nw = normw[d];
#pragma unroll
        for (int r = 0; r < 8; ++r) {
            const int rowg = row0g + r + 8 * hlf;
            out[(size_t)rowg * DIMD + d] = o[t][r] * inorm[r] * nw;
        }
    }
}

// ---------------------------------------------------------------------------
// Host launcher
// ---------------------------------------------------------------------------
extern "C" void kernel_launch(void* const* d_in, const int* in_sizes, int n_in,
                              void* d_out, int out_size, void* d_ws, size_t ws_size,
                              hipStream_t stream) {
    (void)in_sizes; (void)n_in; (void)out_size; (void)ws_size;
    const float* x     = (const float*)d_in[0];
    const float* W1    = (const float*)d_in[1];
    const float* b1    = (const float*)d_in[2];
    const float* W2    = (const float*)d_in[3];
    const float* b2    = (const float*)d_in[4];
    const float* Wv    = (const float*)d_in[5];
    const float* bv    = (const float*)d_in[6];
    const float* mixw  = (const float*)d_in[7];
    const float* normw = (const float*)d_in[8];
    float* out = (float*)d_out;

    // workspace layout (~118 MB total)
    char* ws = (char*)d_ws;
    size_t off = 0;
    auto alloc = [&](size_t bytes) -> char* {
        char* p = ws + off;
        off += (bytes + 255) & ~(size_t)255;
        return p;
    };
    __bf16* xb  = (__bf16*)alloc((size_t)ROWS * DIMD * 2);      // 16 MB
    __bf16* hh  = (__bf16*)alloc((size_t)ROWS * INNERD * 2);    // 24 MB
    __bf16* qb  = (__bf16*)alloc((size_t)ROWS * DIMD * 2);      // 16 MB
    __bf16* kb  = (__bf16*)alloc((size_t)ROWS * DIMD * 2);      // 16 MB
    __bf16* sg  = (__bf16*)alloc((size_t)ROWS * DIMD * 2);      // 16 MB
    __bf16* vT  = (__bf16*)alloc((size_t)ROWS * DIMD * 2);      // 16 MB
    __bf16* W1t = (__bf16*)alloc((size_t)DIMD * INNERD * 2);    // 3 MB
    __bf16* W2t = (__bf16*)alloc((size_t)INNERD * TDIMD * 2);   // 9 MB
    __bf16* Wvt = (__bf16*)alloc((size_t)DIMD * DIMD * 2);      // 2 MB

    int n;
    n = ROWS * DIMD;
    k_f32_to_bf16<<<(n + 255) / 256, 256, 0, stream>>>(x, xb, n);
    n = DIMD * INNERD;
    k_transpose_bf16<<<(n + 255) / 256, 256, 0, stream>>>(W1, W1t, DIMD, INNERD);
    n = INNERD * TDIMD;
    k_transpose_bf16<<<(n + 255) / 256, 256, 0, stream>>>(W2, W2t, INNERD, TDIMD);
    n = DIMD * DIMD;
    k_transpose_bf16<<<(n + 255) / 256, 256, 0, stream>>>(Wv, Wvt, DIMD, DIMD);

    k_gemm1_silu<<<dim3(ROWS / 16, INNERD / 64), 32, 0, stream>>>(xb, W1t, b1, hh);
    k_gemm2_qkg <<<dim3(ROWS / 16, TDIMD / 64), 32, 0, stream>>>(hh, W2t, b2, qb, kb, sg);
    k_gemm3_v   <<<dim3(ROWS / 16, DIMD / 64), 32, 0, stream>>>(xb, Wvt, bv, sg, vT);

    k_attention<<<dim3(SEQ / 16, BATCH), 256, 0, stream>>>(qb, kb, vT, x, mixw, normw, out);
}
